// Model_39676907884608
// MI455X (gfx1250) — matrix operand order repaired and checked by reading, not compiled, not run on hardware
//
#include <hip/hip_runtime.h>
#include <hip/hip_bf16.h>
#include <math.h>

// Problem constants (match reference)
#define BB   2
#define SQ   4096
#define DD   512
#define HH   4

// LDS score row stride in halves: rows stay 16B-aligned; per-lane ds_load_b128
// of A-fragments is conflict-minimal.
#define STRH (SQ + 8)   // 4104 halves per row

typedef __attribute__((ext_vector_type(16))) _Float16 v16h;
typedef __attribute__((ext_vector_type(8)))  _Float16 v8h;
typedef __attribute__((ext_vector_type(4)))  _Float16 v4h;
typedef __attribute__((ext_vector_type(8)))  float    v8f;
typedef __attribute__((ext_vector_type(4)))  float    v4f;

union U16h { v16h v; v8h h[2]; _Float16 e[16]; };

static __device__ __forceinline__ v8f wmma_f16(const v16h& a, const v16h& b, const v8f& c) {
    return __builtin_amdgcn_wmma_f32_16x16x32_f16(false, a, false, b, (short)0, c, false, false);
}

// ---------------------------------------------------------------------------
// Fused attention: one block = 16 query rows, exact softmax over S=4096 keys.
// LDS: 16 x 4104 f16 scores (~128 KB) + 16 f32 row scales.
// Phase 1: Q strip register-resident (16 A-frags = 128 VGPRs); 8 waves each
//          compute scores for 512 keys via WMMA, 2 key tiles per iteration
//          (2 independent accumulator chains) -> f16 in LDS.
// Phase 2: rowwise softmax on f16 in LDS (16 threads/row, shfl reductions),
//          exp written back in place (same element index -> no aliasing).
// Phase 3: each wave owns 64 of the D=512 output cols; kk unrolled x2 with
//          independent P fragments (two aligned ds_load_b128 each); B-frags
//          contiguous from transposed V. Output replicated to H=4 heads.
// ---------------------------------------------------------------------------
__global__ __launch_bounds__(256)
void attn_kernel(const _Float16* __restrict__ Qh, const _Float16* __restrict__ Kh,
                 const _Float16* __restrict__ Vt, float* __restrict__ outp)
{
    extern __shared__ char smem[];
    _Float16* sc = (_Float16*)smem;                        // [16][STRH] f16
    float* invl  = (float*)(smem + (size_t)16 * STRH * 2); // [16] f32

    const int b     = blockIdx.y;
    const int row0  = blockIdx.x * 16;
    const int wave  = threadIdx.x >> 5;
    const int lane  = threadIdx.x & 31;
    const int lhalf = lane >> 4;
    const int l16   = lane & 15;

    // ---- Preload Q strip into registers: 16 A-fragments (scale folded in) ----
    const _Float16* qrow = Qh + ((size_t)b * SQ + row0 + l16) * DD;
    U16h qa[16];
#pragma unroll
    for (int kc = 0; kc < 16; ++kc) {
        const _Float16* ap = qrow + kc * 32 + lhalf * 16;
        qa[kc].h[0] = *(const v8h*)ap;
        qa[kc].h[1] = *(const v8h*)(ap + 8);
    }

    // ---- Phase 1: scores for keys [wave*512, wave*512+512) -> LDS (f16) ----
    for (int kt = 0; kt < 16; ++kt) {
        const int key0 = wave * 512 + kt * 32;   // two 16-key tiles
        const _Float16* krow0 =
            Kh + ((size_t)b * SQ + key0 + l16) * DD + lhalf * 16;
        const _Float16* krow1 = krow0 + 16 * DD;
        if (kt + 1 < 16) {   // prefetch next iteration's K rows (global_prefetch_b8)
            __builtin_prefetch(krow0 + 32 * DD, 0, 0);
            __builtin_prefetch(krow1 + 32 * DD, 0, 0);
        }
        v8f c0 = {}, c1 = {};
#pragma unroll
        for (int kc = 0; kc < 16; ++kc) {
            v16h b0 = *(const v16h*)(krow0 + kc * 32);
            v16h b1 = *(const v16h*)(krow1 + kc * 32);
            c0 = wmma_f16(qa[kc].v, b0, c0);
            c1 = wmma_f16(qa[kc].v, b1, c1);
        }
#pragma unroll
        for (int r = 0; r < 8; ++r) {
            int m = r + lhalf * 8;
            sc[m * STRH + key0 + l16]      = (_Float16)c0[r];
            sc[m * STRH + key0 + 16 + l16] = (_Float16)c1[r];
        }
    }
    __syncthreads();

    // ---- Phase 2: exact softmax, 16 threads per row, f16 in place ----
    {
        const int row = threadIdx.x >> 4;
        const int c0  = threadIdx.x & 15;
        _Float16* srow = sc + row * STRH;
        float mx = -3.0e38f;
        for (int j = c0; j < SQ; j += 16)
            mx = fmaxf(mx, (float)srow[j]);
        for (int off = 8; off >= 1; off >>= 1)
            mx = fmaxf(mx, __shfl_xor(mx, off, 16));
        float sum = 0.f;
        for (int j = c0; j < SQ; j += 16) {
            float e = __expf((float)srow[j] - mx);
            srow[j] = (_Float16)e;    // same element index: no cross-thread alias
            sum += e;
        }
        for (int off = 8; off >= 1; off >>= 1)
            sum += __shfl_xor(sum, off, 16);
        if (c0 == 0) invl[row] = 1.0f / sum;
    }
    __syncthreads();

    // ---- Phase 3: O = P @ V over all 4096 keys; wave owns cols [col0,col0+64) ----
    const int col0 = wave * 64;
    const _Float16* prow = sc + l16 * STRH;
    const _Float16* vbase =
        Vt + ((size_t)b * DD + col0 + l16) * SQ + lhalf * 16;
    v8f o[4] = {};
    for (int kk = 0; kk < SQ / 32; kk += 2) {
        U16h a0, a1;   // two independent P fragments (aligned 16B LDS loads)
        const _Float16* pp = prow + kk * 32 + lhalf * 16;
        a0.h[0] = *(const v8h*)pp;
        a0.h[1] = *(const v8h*)(pp + 8);
        a1.h[0] = *(const v8h*)(pp + 32);
        a1.h[1] = *(const v8h*)(pp + 40);
#pragma unroll
        for (int t = 0; t < 4; ++t) {
            const _Float16* bp = vbase + (size_t)t * 16 * SQ + kk * 32;
            v16h b0 = *(const v16h*)bp;
            v16h b1 = *(const v16h*)(bp + 32);
            o[t] = wmma_f16(a0.v, b0, o[t]);
            o[t] = wmma_f16(a1.v, b1, o[t]);
        }
    }

#pragma unroll
    for (int t = 0; t < 4; ++t) {
#pragma unroll
        for (int r = 0; r < 8; ++r) {
            int m = r + lhalf * 8;
            int n = col0 + t * 16 + l16;
            float val = o[t][r] * invl[m];
            size_t base = ((size_t)b * HH * SQ + (size_t)(row0 + m)) * DD + n;
#pragma unroll
            for (int h = 0; h < HH; ++h)
                outp[base + (size_t)h * SQ * DD] = val;   // identical across heads
        }
    }
}

// ---------------------------------------------------------------------------
// f32 -> f16 cast, 4 elements per thread
// ---------------------------------------------------------------------------
__global__ void cast_f16_kernel(const float* __restrict__ src,
                                _Float16* __restrict__ dst, int n4) {
    int i = blockIdx.x * blockDim.x + threadIdx.x;
    if (i < n4) {
        v4f a = ((const v4f*)src)[i];
        v4h o;
        o[0] = (_Float16)a[0];
        o[1] = (_Float16)a[1];
        o[2] = (_Float16)a[2];
        o[3] = (_Float16)a[3];
        ((v4h*)dst)[i] = o;
    }
}

// ---------------------------------------------------------------------------
// WMMA projection: out = (ah @ wh^T + bias) * scale, f16 in / f16 out.
// Block: 256 threads = 8 waves; 16 rows x full D=512 per block.
// transposeOut=1 stores out[b][d][s] (for V) instead of out[b][s][d].
// ---------------------------------------------------------------------------
__global__ __launch_bounds__(256)
void proj_kernel(const _Float16* __restrict__ ah, const _Float16* __restrict__ wh,
                 const float* __restrict__ bias, _Float16* __restrict__ outp,
                 float scale, int transposeOut)
{
    const int b     = blockIdx.y;
    const int row0  = blockIdx.x * 16;
    const int wave  = threadIdx.x >> 5;
    const int lane  = threadIdx.x & 31;
    const int lhalf = lane >> 4;
    const int l16   = lane & 15;
    const int col0  = wave * 64;

    v8f acc[4] = {};
    const _Float16* arow = ah + ((size_t)b * SQ + (row0 + l16)) * DD;

    for (int kc = 0; kc < DD / 32; ++kc) {
        U16h a;
        const _Float16* ap = arow + kc * 32 + lhalf * 16;
        a.h[0] = *(const v8h*)ap;
        a.h[1] = *(const v8h*)(ap + 8);
#pragma unroll
        for (int t = 0; t < 4; ++t) {
            const _Float16* bp =
                wh + (size_t)(col0 + t * 16 + l16) * DD + kc * 32 + lhalf * 16;
            acc[t] = wmma_f16(a.v, *(const v16h*)bp, acc[t]);
        }
    }

#pragma unroll
    for (int t = 0; t < 4; ++t) {
#pragma unroll
        for (int r = 0; r < 8; ++r) {
            int m = r + lhalf * 8;
            int n = col0 + t * 16 + l16;
            float val = (acc[t][r] + bias[n]) * scale;
            if (!transposeOut)
                outp[((size_t)b * SQ + row0 + m) * DD + n] = (_Float16)val;
            else
                outp[((size_t)b * DD + n) * SQ + row0 + m] = (_Float16)val;
        }
    }
}

// ---------------------------------------------------------------------------
// Host launcher
// ---------------------------------------------------------------------------
extern "C" void kernel_launch(void* const* d_in, const int* in_sizes, int n_in,
                              void* d_out, int out_size, void* d_ws, size_t ws_size,
                              hipStream_t stream)
{
    const float* q  = (const float*)d_in[0];
    const float* k  = (const float*)d_in[1];
    const float* v  = (const float*)d_in[2];
    const float* Wq = (const float*)d_in[3];
    const float* bq = (const float*)d_in[4];
    const float* Wv = (const float*)d_in[5];
    const float* bv = (const float*)d_in[6];
    float* outp = (float*)d_out;

    // Workspace layout (halves). Total ~21.5M halves ~= 43 MB.
    _Float16* ws = (_Float16*)d_ws;
    const size_t nQKV = (size_t)BB * SQ * DD;   // 4,194,304
    const size_t nW   = (size_t)DD * DD;        // 262,144
    _Float16* qh  = ws;                 // f16(q)
    _Float16* kh  = qh  + nQKV;         // f16(k)   (used raw for scores)
    _Float16* vh  = kh  + nQKV;         // f16(v)
    _Float16* wqh = vh  + nQKV;         // f16(Wq)
    _Float16* wvh = wqh + nW;           // f16(Wv)
    _Float16* Qp  = wvh + nW;           // f16((q@Wq^T + bq)/sqrt(D))  [B][S][D]
    _Float16* Vt  = Qp  + nQKV;         // f16(v@Wv^T + bv) transposed [B][D][S]

    // ---- casts to f16 ----
    {
        int n4;
        n4 = (int)(nQKV / 4);
        cast_f16_kernel<<<dim3((n4 + 255) / 256), dim3(256), 0, stream>>>(q,  qh, n4);
        cast_f16_kernel<<<dim3((n4 + 255) / 256), dim3(256), 0, stream>>>(k,  kh, n4);
        cast_f16_kernel<<<dim3((n4 + 255) / 256), dim3(256), 0, stream>>>(v,  vh, n4);
        n4 = (int)(nW / 4);
        cast_f16_kernel<<<dim3((n4 + 255) / 256), dim3(256), 0, stream>>>(Wq, wqh, n4);
        cast_f16_kernel<<<dim3((n4 + 255) / 256), dim3(256), 0, stream>>>(Wv, wvh, n4);
    }

    // ---- projections (WMMA) ----
    const float qscale = 0.044194173824159216f;   // 1/sqrt(512)
    proj_kernel<<<dim3(SQ / 16, BB), dim3(256), 0, stream>>>(qh, wqh, bq, Qp, qscale, 0);
    proj_kernel<<<dim3(SQ / 16, BB), dim3(256), 0, stream>>>(vh, wvh, bv, Vt, 1.0f, 1);

    // ---- fused attention ----
    const size_t shmem = (size_t)16 * STRH * 2 + 16 * sizeof(float); // ~128.3 KB
    (void)hipFuncSetAttribute(reinterpret_cast<const void*>(attn_kernel),
                              hipFuncAttributeMaxDynamicSharedMemorySize,
                              (int)shmem);
    attn_kernel<<<dim3(SQ / 16, BB), dim3(256), shmem, stream>>>(Qp, kh, Vt, outp);
}
